// SeqIVModel_50723563766203
// MI455X (gfx1250) — matrix operand order repaired and checked by reading, not compiled, not run on hardware
//
#include <hip/hip_runtime.h>

// ---------------------------------------------------------------------------
// Autoregressive LSTM (B=2048, P=16, H=512, T=100) for MI455X (gfx1250).
// One workgroup owns a 16-row batch tile and iterates all 100 timesteps.
// Gates GEMM (16 x 2048 x 512 per step) runs on v_wmma_f32_16x16x32_bf16.
// ---------------------------------------------------------------------------

#define BATCH   2048
#define PDIM    16
#define HDIM    512
#define GDIM    2048        // 4*H
#define TSTEPS  100
#define HPAD    520         // padded bf16 row (1040B) -> 4-bank rotation/row
#define THREADS 512         // 16 waves
#define ROWS    16          // batch rows per workgroup

typedef __attribute__((ext_vector_type(16))) __bf16 v16bf;
typedef __attribute__((ext_vector_type(8)))  float  v8f;

union FragBF {
  uint4 q[2];
  v16bf v;
};

__device__ __forceinline__ unsigned short f2bf(float f) {
  unsigned int u = __float_as_uint(f);
  u += 0x7fffu + ((u >> 16) & 1u);          // round-to-nearest-even
  return (unsigned short)(u >> 16);
}

__device__ __forceinline__ float fsigmoid(float x) {
  return __builtin_amdgcn_rcpf(1.0f + __expf(-x));
}
__device__ __forceinline__ float ftanh(float x) {
  // tanh(x) = 1 - 2/(exp(2x)+1); saturates correctly at +-1
  return 1.0f - 2.0f * __builtin_amdgcn_rcpf(1.0f + __expf(2.0f * x));
}

// Convert W_hh to bf16 (row-major, L2-resident) and fuse b_ih + b_hh.
__global__ __launch_bounds__(256) void prep_kernel(
    const float* __restrict__ Whh, const float* __restrict__ b_ih,
    const float* __restrict__ b_hh, unsigned short* __restrict__ whh_bf,
    float* __restrict__ bias) {
  const int i = blockIdx.x * 256 + threadIdx.x;
  if (i < GDIM * HDIM) whh_bf[i] = f2bf(Whh[i]);
  if (i < GDIM) bias[i] = b_ih[i] + b_hh[i];
}

__global__ __launch_bounds__(THREADS) void lstm_kernel(
    const float* __restrict__ physical, const float* __restrict__ W_pe,
    const float* __restrict__ b_pe, const float* __restrict__ W_ih,
    const float* __restrict__ W_fc, const float* __restrict__ b_fc,
    const float* __restrict__ init_input,
    const unsigned short* __restrict__ whh_bf, const float* __restrict__ bias,
    float* __restrict__ out) {
  // Ping-pong h buffer (bf16) + ping-pong autoregressive scalar buffer.
  __shared__ __align__(16) unsigned short hbuf[2][ROWS][HPAD];
  __shared__ float predBuf[2][ROWS];

  const int tid   = threadIdx.x;
  const int wave  = tid >> 5;
  const int lane  = tid & 31;
  const int lo    = lane & 15;       // N / M index within a 16-wide tile
  const int hi    = lane >> 4;       // half-wave selector
  const int mbase = blockIdx.x * ROWS;

  // ---- h0 = tanh(physical @ W_pe^T + b_pe); c0 = 0; x0 = init_input -------
  for (int idx = tid; idx < ROWS * HDIM; idx += THREADS) {
    const int m = idx >> 9;
    const int j = idx & (HDIM - 1);
    float acc = b_pe[j];
    const float* wrow = W_pe + j * PDIM;
    const float* prow = physical + (mbase + m) * PDIM;
#pragma unroll
    for (int p = 0; p < PDIM; ++p) acc += prow[p] * wrow[p];
    hbuf[0][m][j] = f2bf(ftanh(acc));
  }
  if (tid < ROWS) predBuf[0][tid] = init_input[0];

  // Persistent cell state in accumulator layout: cst[s][r] -> row r+8*hi,
  // hidden column nbase + s*16 + lo.
  v8f cst[2];
#pragma unroll
  for (int s = 0; s < 2; ++s)
#pragma unroll
    for (int r = 0; r < 8; ++r) cst[s][r] = 0.0f;

  __syncthreads();

  const int nbase = wave * 32;   // this wave's hidden-index range [nbase, nbase+32)

  for (int t = 0; t < TSTEPS; ++t) {
    const int cur = t & 1;
    const int nxt = cur ^ 1;
    if (tid < ROWS) predBuf[nxt][tid] = 0.0f;
    __syncthreads();

    // Autoregressive input x[m] for my 8 rows.
    float xv[8];
#pragma unroll
    for (int r = 0; r < 8; ++r) xv[r] = predBuf[cur][r + 8 * hi];

    float predPart[8];
#pragma unroll
    for (int r = 0; r < 8; ++r) predPart[r] = 0.0f;

    // A-fragment source: LDS row m = lo (same for both half-waves).
    const unsigned short* hrow = &hbuf[cur][lo][0];

#pragma unroll
    for (int s = 0; s < 2; ++s) {
      const int ncol = nbase + s * 16 + lo;            // hidden column
      // Four gate tiles (i,f,g,o) sharing hidden column ncol.
      v8f acc[4];
#pragma unroll
      for (int g = 0; g < 4; ++g) {
        const int gcol  = g * HDIM + ncol;
        const float bb  = bias[gcol];
        const float wih = W_ih[gcol];
#pragma unroll
        for (int r = 0; r < 8; ++r) acc[g][r] = bb + wih * xv[r];  // bias + x*W_ih
      }

      // K reduction: 16 x v_wmma_f32_16x16x32_bf16 per gate tile.
#pragma unroll 2
      for (int k0 = 0; k0 < HDIM; k0 += 32) {
        // A layout (16-bit 16x32): lane<16 -> K {k0..k0+7, k0+16..k0+23},
        // lane>=16 -> +8. Two ds_load_b128.
        FragBF a;
        a.q[0] = *reinterpret_cast<const uint4*>(hrow + k0 + 8 * hi);
        a.q[1] = *reinterpret_cast<const uint4*>(hrow + k0 + 16 + 8 * hi);
#pragma unroll
        for (int g = 0; g < 4; ++g) {
          // B layout (16-bit KxN): lane = column N = gate row of W_hh;
          // K {k0..k0+7, k0+16..k0+23} per lane, half-waves +8 (matches A).
          const unsigned short* brow =
              whh_bf + (size_t)(g * HDIM + ncol) * HDIM + k0 + 8 * hi;
          FragBF b;
          b.q[0] = *reinterpret_cast<const uint4*>(brow);
          b.q[1] = *reinterpret_cast<const uint4*>(brow + 16);
          acc[g] = __builtin_amdgcn_wmma_f32_16x16x32_bf16(
              false, a.v, false, b.v, (short)0, acc[g], false, false);
        }
      }

      // Elementwise LSTM cell (local to this wave's hidden range).
      const float wfc = W_fc[ncol];
#pragma unroll
      for (int r = 0; r < 8; ++r) {
        const float ig = fsigmoid(acc[0][r]);
        const float fg = fsigmoid(acc[1][r]);
        const float gg = ftanh(acc[2][r]);
        const float og = fsigmoid(acc[3][r]);
        const float cn = fg * cst[s][r] + ig * gg;
        cst[s][r] = cn;
        const float hn = og * ftanh(cn);
        hbuf[nxt][r + 8 * hi][ncol] = f2bf(hn);       // next step's A matrix
        predPart[r] += hn * wfc;                      // FC partial
      }
    }

    // Reduce FC partials across each 16-lane half (same row m, different n),
    // then one ds_add_f32 per (wave, row).
#pragma unroll
    for (int r = 0; r < 8; ++r) {
      float v = predPart[r];
      v += __shfl_xor(v, 1);
      v += __shfl_xor(v, 2);
      v += __shfl_xor(v, 4);
      v += __shfl_xor(v, 8);
      if (lo == 0) atomicAdd(&predBuf[nxt][r + 8 * hi], v);
    }
    __syncthreads();

    if (tid < ROWS) {
      const float p = predBuf[nxt][tid] + b_fc[0];
      predBuf[nxt][tid] = p;                          // x for step t+1
      out[(size_t)(mbase + tid) * TSTEPS + t] = p;    // out[b, t]
    }
    // next iteration's first barrier orders these writes vs. readers
  }
}

extern "C" void kernel_launch(void* const* d_in, const int* in_sizes, int n_in,
                              void* d_out, int out_size, void* d_ws, size_t ws_size,
                              hipStream_t stream) {
  (void)in_sizes; (void)n_in; (void)out_size; (void)ws_size;
  const float* physical   = (const float*)d_in[0];
  const float* W_pe       = (const float*)d_in[1];
  const float* b_pe       = (const float*)d_in[2];
  const float* W_ih       = (const float*)d_in[3];
  const float* W_hh       = (const float*)d_in[4];
  const float* b_ih       = (const float*)d_in[5];
  const float* b_hh       = (const float*)d_in[6];
  const float* W_fc       = (const float*)d_in[7];
  const float* b_fc       = (const float*)d_in[8];
  const float* init_input = (const float*)d_in[9];
  // d_in[10] = max_len (hardcoded TSTEPS=100)

  unsigned short* whh_bf = (unsigned short*)d_ws;
  float* bias = (float*)((char*)d_ws + (size_t)GDIM * HDIM * sizeof(unsigned short));

  prep_kernel<<<(GDIM * HDIM + 255) / 256, 256, 0, stream>>>(
      W_hh, b_ih, b_hh, whh_bf, bias);

  lstm_kernel<<<BATCH / ROWS, THREADS, 0, stream>>>(
      physical, W_pe, b_pe, W_ih, W_fc, b_fc, init_input,
      whh_bf, bias, (float*)d_out);
}
